// TinyRNNRegressor_30571577213410
// MI455X (gfx1250) — compile-verified, matrix operand order checked by reading, not run on hardware
//
#include <hip/hip_runtime.h>
#include <hip/hip_bf16.h>

typedef __attribute__((ext_vector_type(16))) __bf16 v16bf;
typedef __attribute__((ext_vector_type(8)))  float  v8f;

#define T_LEN 2048
#define H_DIM 128
#define G_DIM 384
#define B_DIM 64

__device__ inline v16bf cvt16(const float* __restrict__ p) {
  v16bf r;
  #pragma unroll
  for (int i = 0; i < 16; ++i) r[i] = (__bf16)p[i];
  return r;
}

__device__ inline void store_bf16x4(__bf16* dst, float a, float b, float c, float d) {
  union { __bf16 h[4]; unsigned long long u; } pk;
  pk.h[0] = (__bf16)a; pk.h[1] = (__bf16)b;
  pk.h[2] = (__bf16)c; pk.h[3] = (__bf16)d;
  *reinterpret_cast<unsigned long long*>(dst) = pk.u;  // 8B aligned by construction
}

__device__ inline float fast_sigmoid(float x) {
  return 1.0f / (1.0f + __expf(-x));
}
__device__ inline float fast_tanh(float x) {
  float e = __expf(-2.0f * x);
  return (1.0f - e) / (1.0f + e);
}

// Persistent GRU scan over T for 16 batch rows (one workgroup, 24 waves).
// Wave w owns gate columns [16w, 16w+16): waves 0..7 -> r, 8..15 -> z, 16..23 -> n.
// Per step: gates[16,384] = [h(128) | x_t(IN_K)] @ [W_hh | W_ih]^T via WMMA bf16;
// n-gate keeps hn and xn in separate accumulators so n = tanh(xn + r*hn).
// Waves 0..15 do the gate nonlinearity (wave m owns batch row m, h kept in regs);
// waves 16..23 concurrently stage prefetched x_{t+1} into LDS and prefetch x_{t+2}.
template<int IN_K, bool HEAD>
__global__ __launch_bounds__(768, 1)
void gru_scan_kernel(const float* __restrict__ xin,    // [B, T, IN_K]
                     const float* __restrict__ W_ih,   // [384, IN_K]
                     const float* __restrict__ W_hh,   // [384, 128]
                     const float* __restrict__ b_ih,   // [384]
                     const float* __restrict__ b_hh,   // [384]
                     const float* __restrict__ W_head, // [128] (HEAD only)
                     float* __restrict__ hout,         // [B, T, 128] (!HEAD)
                     float* __restrict__ out)          // [B, T]      (HEAD)
{
  constexpr int KROW = H_DIM + IN_K;      // A-row: [h | x_t]
  constexpr int NCHX = IN_K / 32;         // x K-chunks
  constexpr int QPR  = IN_K / 4;          // float4s per staged row
  constexpr int XV   = (16 * QPR) / 256;  // float4s per staging thread (1 or 2)

  __shared__ __align__(32) __bf16 A_lds[16 * KROW];    // WMMA A source (bf16)
  __shared__ __align__(16) float  g_lds[16 * G_DIM];   // r | z | hn
  __shared__ __align__(16) float  g2_lds[16 * H_DIM];  // xn

  const int tid  = threadIdx.x;
  const int wv   = tid >> 5;
  const int lane = tid & 31;
  const int b0   = blockIdx.x * 16;

  // ---- weight fragments as WMMA B operands (K x N = 32x16 chunks), in VGPRs ----
  // B[k][n] = W[n][k]; lane layout: N = lane&15, K-half = (lane<16)?0:16.
  const int n0   = wv * 16;
  const int ncol = n0 + (lane & 15);
  const int kh   = (lane < 16) ? 0 : 16;

  v16bf Bh[4];
  #pragma unroll
  for (int c = 0; c < 4; ++c)
    Bh[c] = cvt16(&W_hh[(size_t)ncol * H_DIM + c * 32 + kh]);
  v16bf Bx[NCHX];
  #pragma unroll
  for (int c = 0; c < NCHX; ++c)
    Bx[c] = cvt16(&W_ih[(size_t)ncol * IN_K + c * 32 + kh]);

  // ---- per-thread loop-invariant state ----
  // Gate threads (waves 0..15): biases, head weights, h state in registers.
  const int gm = tid >> 5;          // batch row owned (gate threads)
  const int j0 = (lane) * 4;        // 4 h-columns owned
  float br[4], bz[4], bhn[4], bxn[4], wh[4];
  float hreg[4] = {0.f, 0.f, 0.f, 0.f};
  if (tid < 512) {
    #pragma unroll
    for (int jj = 0; jj < 4; ++jj) {
      br[jj]  = b_ih[j0 + jj] + b_hh[j0 + jj];
      bz[jj]  = b_ih[128 + j0 + jj] + b_hh[128 + j0 + jj];
      bhn[jj] = b_hh[256 + j0 + jj];
      bxn[jj] = b_ih[256 + j0 + jj];
      wh[jj]  = HEAD ? W_head[j0 + jj] : 0.f;
    }
  }

  // Staging threads (waves 16..23): prefetch registers for x.
  const int st = tid - 512;
  float4 xpre[XV];

  // ---- prologue: zero h-part of A; stage x_0; prefetch x_1 ----
  for (int i = tid; i < 16 * H_DIM; i += 768) A_lds[i] = (__bf16)0.0f;
  if (tid >= 512) {
    #pragma unroll
    for (int k = 0; k < XV; ++k) {
      int e = st + 256 * k;
      int m = e / QPR, q = e - m * QPR;
      float4 v = *reinterpret_cast<const float4*>(
          &xin[((size_t)(b0 + m) * T_LEN + 0) * IN_K + q * 4]);
      store_bf16x4(&A_lds[m * KROW + H_DIM + q * 4], v.x, v.y, v.z, v.w);
      xpre[k] = *reinterpret_cast<const float4*>(
          &xin[((size_t)(b0 + m) * T_LEN + 1) * IN_K + q * 4]);
    }
  }
  __syncthreads();

  const int am = lane & 15;  // A-fragment row
  const __bf16* Arow = &A_lds[am * KROW + kh];
  const int crow = (lane >= 16) ? 8 : 0;  // C-tile row offset for this lane half

  for (int t = 0; t < T_LEN; ++t) {
    // ---- GEMM step: (4 + NCHX) k-chunk WMMAs per wave ----
    v8f acc = {};  // h-part accumulation
    #pragma unroll
    for (int c = 0; c < 4; ++c) {
      v16bf a = *reinterpret_cast<const v16bf*>(Arow + c * 32);
      acc = __builtin_amdgcn_wmma_f32_16x16x32_bf16(false, a, false, Bh[c],
                                                    (short)0, acc, false, false);
    }
    if (wv < 16) {
      // r/z gates: x-part fuses into the same accumulator
      #pragma unroll
      for (int c = 0; c < NCHX; ++c) {
        v16bf a = *reinterpret_cast<const v16bf*>(Arow + H_DIM + c * 32);
        acc = __builtin_amdgcn_wmma_f32_16x16x32_bf16(false, a, false, Bx[c],
                                                      (short)0, acc, false, false);
      }
      #pragma unroll
      for (int i = 0; i < 8; ++i)
        g_lds[(i + crow) * G_DIM + ncol] = acc[i];
    } else {
      // n gate: hn and xn stay separate (n = tanh(xn + r*hn))
      v8f accx = {};
      #pragma unroll
      for (int c = 0; c < NCHX; ++c) {
        v16bf a = *reinterpret_cast<const v16bf*>(Arow + H_DIM + c * 32);
        accx = __builtin_amdgcn_wmma_f32_16x16x32_bf16(false, a, false, Bx[c],
                                                       (short)0, accx, false, false);
      }
      #pragma unroll
      for (int i = 0; i < 8; ++i) {
        g_lds[(i + crow) * G_DIM + ncol]          = acc[i];   // hn
        g2_lds[(i + crow) * H_DIM + (ncol - 256)] = accx[i];  // xn
      }
    }
    __syncthreads();

    if (tid < 512) {
      // ---- gates + state update: wave m owns batch row m, 4 cols/lane ----
      const float4 gr = *reinterpret_cast<const float4*>(&g_lds[gm * G_DIM + j0]);
      const float4 gz = *reinterpret_cast<const float4*>(&g_lds[gm * G_DIM + 128 + j0]);
      const float4 gn = *reinterpret_cast<const float4*>(&g_lds[gm * G_DIM + 256 + j0]);
      const float4 gx = *reinterpret_cast<const float4*>(&g2_lds[gm * H_DIM + j0]);
      const float grr[4] = {gr.x, gr.y, gr.z, gr.w};
      const float gzz[4] = {gz.x, gz.y, gz.z, gz.w};
      const float gnn[4] = {gn.x, gn.y, gn.z, gn.w};
      const float gxx[4] = {gx.x, gx.y, gx.z, gx.w};
      float part = 0.0f;
      #pragma unroll
      for (int jj = 0; jj < 4; ++jj) {
        float r    = fast_sigmoid(grr[jj] + br[jj]);
        float z    = fast_sigmoid(gzz[jj] + bz[jj]);
        float n    = fast_tanh(gxx[jj] + bxn[jj] + r * (gnn[jj] + bhn[jj]));
        hreg[jj]   = (1.0f - z) * n + z * hreg[jj];
        part      += hreg[jj] * wh[jj];
      }
      store_bf16x4(&A_lds[gm * KROW + j0], hreg[0], hreg[1], hreg[2], hreg[3]);
      if constexpr (!HEAD) {
        float4 hv = {hreg[0], hreg[1], hreg[2], hreg[3]};
        *reinterpret_cast<float4*>(
            &hout[((size_t)(b0 + gm) * T_LEN + t) * H_DIM + j0]) = hv;
      } else {
        #pragma unroll
        for (int off = 16; off > 0; off >>= 1)
          part += __shfl_xor(part, off, 32);
        if (lane == 0) out[(size_t)(b0 + gm) * T_LEN + t] = part;
      }
    } else {
      // ---- staging: write prefetched x_{t+1}; prefetch x_{t+2} ----
      if (t + 1 < T_LEN) {
        #pragma unroll
        for (int k = 0; k < XV; ++k) {
          int e = st + 256 * k;
          int m = e / QPR, q = e - m * QPR;
          store_bf16x4(&A_lds[m * KROW + H_DIM + q * 4],
                       xpre[k].x, xpre[k].y, xpre[k].z, xpre[k].w);
        }
      }
      if (t + 2 < T_LEN) {
        #pragma unroll
        for (int k = 0; k < XV; ++k) {
          int e = st + 256 * k;
          int m = e / QPR, q = e - m * QPR;
          xpre[k] = *reinterpret_cast<const float4*>(
              &xin[((size_t)(b0 + m) * T_LEN + (t + 2)) * IN_K + q * 4]);
        }
      }
    }
    __syncthreads();
  }
}

extern "C" void kernel_launch(void* const* d_in, const int* in_sizes, int n_in,
                              void* d_out, int out_size, void* d_ws, size_t ws_size,
                              hipStream_t stream) {
  const float* x      = (const float*)d_in[0];
  const float* W_ih0  = (const float*)d_in[1];
  const float* W_hh0  = (const float*)d_in[2];
  const float* b_ih0  = (const float*)d_in[3];
  const float* b_hh0  = (const float*)d_in[4];
  const float* W_ih1  = (const float*)d_in[5];
  const float* W_hh1  = (const float*)d_in[6];
  const float* b_ih1  = (const float*)d_in[7];
  const float* b_hh1  = (const float*)d_in[8];
  const float* W_head = (const float*)d_in[9];
  float* out = (float*)d_out;
  float* h0  = (float*)d_ws;  // [B, T, H] f32 = 64 MiB scratch

  dim3 grid(B_DIM / 16), block(768);
  gru_scan_kernel<64, false><<<grid, block, 0, stream>>>(
      x, W_ih0, W_hh0, b_ih0, b_hh0, nullptr, h0, nullptr);
  gru_scan_kernel<128, true><<<grid, block, 0, stream>>>(
      h0, W_ih1, W_hh1, b_ih1, b_hh1, W_head, nullptr, out);
}
